// BGraphAttentionLayer_59072980189840
// MI455X (gfx1250) — matrix operand order repaired and checked by reading, not compiled, not run on hardware
//
#include <hip/hip_runtime.h>
#include <hip/hip_bf16.h>

typedef __attribute__((ext_vector_type(16))) _Float16 v16h;
typedef __attribute__((ext_vector_type(8)))  float    v8f;
typedef unsigned int u32;

#define N_NODES 8192
#define F_IN    256
#define F_OUT   64
#define ALPHA   0.2f
#define NEG_BIG (-9.0e15f)
#define KCHUNK  64      // K per barrier round = 2 WMMAs
#define WSTRIDE 66      // halves per row in LDS weight/B tiles (33 dwords, odd)

// ---------------------------------------------------------------------------
// Kernel 1: Wh = h @ W  (fp32 accumulate) + f16 copy for the WMMA stage.
// ---------------------------------------------------------------------------
__global__ void wh_kernel(const float* __restrict__ h1, const float* __restrict__ W1,
                          const float* __restrict__ h2, const float* __restrict__ W2,
                          float* __restrict__ Wh1, _Float16* __restrict__ Wh1h,
                          float* __restrict__ Wh2, _Float16* __restrict__ Wh2h) {
  const float* h; const float* W; float* Wh; _Float16* Whh;
  if (blockIdx.y == 0) { h = h1; W = W1; Wh = Wh1; Whh = Wh1h; }
  else                 { h = h2; W = W2; Wh = Wh2; Whh = Wh2h; }
  __shared__ float sh[4 * F_IN];
  const int tid = threadIdx.x;
  const int i0  = blockIdx.x * 4;
  #pragma unroll
  for (int r = 0; r < 4; ++r) {
    int idx = tid + r * 256;
    sh[idx] = h[(size_t)i0 * F_IN + idx];
  }
  __syncthreads();
  const int r = tid >> 6, f = tid & 63;
  float acc = 0.f;
  #pragma unroll 8
  for (int k = 0; k < F_IN; ++k)
    acc = fmaf(sh[r * F_IN + k], W[k * F_OUT + f], acc);
  size_t o = (size_t)(i0 + r) * F_OUT + f;
  Wh[o]  = acc;
  Whh[o] = (_Float16)acc;
}

// ---------------------------------------------------------------------------
// Kernel 2: s1[i] = Wh1[i,:] . a[0:64] ; s2[i] = Wh2[i,:] . a[64:128]
// ---------------------------------------------------------------------------
__global__ void score_kernel(const float* __restrict__ Wh1, const float* __restrict__ Wh2,
                             const float* __restrict__ a,
                             float* __restrict__ s1, float* __restrict__ s2) {
  __shared__ float red[256];
  const int tid = threadIdx.x;
  const int row = blockIdx.x * 4 + (tid >> 6);
  const int f   = tid & 63;
  const float* Wh; const float* av; float* sout; int i;
  if (row < N_NODES) { Wh = Wh1; av = a;         sout = s1; i = row; }
  else               { Wh = Wh2; av = a + F_OUT; sout = s2; i = row - N_NODES; }
  red[tid] = Wh[(size_t)i * F_OUT + f] * av[f];
  __syncthreads();
  #pragma unroll
  for (int off = 32; off > 0; off >>= 1) {
    if ((tid & 63) < off) red[tid] += red[tid + off];
    __syncthreads();
  }
  if ((tid & 63) == 0) sout[i] = red[tid];
}

// ---------------------------------------------------------------------------
// Kernel 3: row softmax stats — one block per row, adj row in registers.
// ---------------------------------------------------------------------------
__global__ void rowstats_kernel(const float* __restrict__ adj, const float* __restrict__ s1,
                                const float* __restrict__ s2,
                                float* __restrict__ rmax, float* __restrict__ rrsum) {
  __shared__ float red[256];
  const int i = blockIdx.x, tid = threadIdx.x;
  const float si = s1[i];
  float v[32];
  float m = NEG_BIG;
  #pragma unroll
  for (int t = 0; t < 32; ++t) {
    int j = tid + t * 256;
    float aij = adj[(size_t)i * N_NODES + j];
    float sc  = si + s2[j];
    sc = sc > 0.f ? sc : ALPHA * sc;
    float att = aij > 0.f ? sc : NEG_BIG;
    v[t] = att;
    m = fmaxf(m, att);
  }
  red[tid] = m; __syncthreads();
  #pragma unroll
  for (int off = 128; off > 0; off >>= 1) {
    if (tid < off) red[tid] = fmaxf(red[tid], red[tid + off]);
    __syncthreads();
  }
  const float M = red[0];
  __syncthreads();
  float s = 0.f;
  #pragma unroll
  for (int t = 0; t < 32; ++t) s += __expf(v[t] - M);
  red[tid] = s; __syncthreads();
  #pragma unroll
  for (int off = 128; off > 0; off >>= 1) {
    if (tid < off) red[tid] += red[tid + off];
    __syncthreads();
  }
  if (tid == 0) { rmax[i] = M; rrsum[i] = 1.f / red[0]; }
}

// ---------------------------------------------------------------------------
// Kernel 4/5: column softmax stats via online-softmax partials + merge.
// ---------------------------------------------------------------------------
__global__ void colpart_kernel(const float* __restrict__ adj, const float* __restrict__ s1,
                               const float* __restrict__ s2,
                               float* __restrict__ partM, float* __restrict__ partS) {
  __shared__ float sh_s1[256];
  const int tid = threadIdx.x;
  const int c   = blockIdx.x * 256 + tid;
  const int r0  = blockIdx.y * 256;
  sh_s1[tid] = s1[r0 + tid];
  __syncthreads();
  const float sj = s2[c];
  float m = NEG_BIG, s = 0.f;
  for (int r = 0; r < 256; ++r) {
    float aij = adj[(size_t)(r0 + r) * N_NODES + c];
    float sc  = sh_s1[r] + sj;
    sc = sc > 0.f ? sc : ALPHA * sc;
    float att = aij > 0.f ? sc : NEG_BIG;
    float nm  = fmaxf(m, att);
    s = s * __expf(m - nm) + __expf(att - nm);
    m = nm;
  }
  partM[(size_t)blockIdx.y * N_NODES + c] = m;
  partS[(size_t)blockIdx.y * N_NODES + c] = s;
}

__global__ void colreduce_kernel(const float* __restrict__ partM, const float* __restrict__ partS,
                                 float* __restrict__ cmax, float* __restrict__ crsum) {
  const int c = blockIdx.x * 256 + threadIdx.x;
  float M = NEG_BIG, S = 0.f;
  #pragma unroll
  for (int g = 0; g < 32; ++g) {
    float pm = partM[(size_t)g * N_NODES + c];
    float ps = partS[(size_t)g * N_NODES + c];
    float nm = fmaxf(M, pm);
    S = S * __expf(M - nm) + ps * __expf(pm - nm);
    M = nm;
  }
  cmax[c] = M; crsum[c] = 1.f / S;
}

// ---------------------------------------------------------------------------
// Pack per-K softmax stats: pack[k] = { s_k, kmax, 1/ksum, 0 } so the hot
// aggregation loop does ONE b128 load instead of three scalar loads.
// ---------------------------------------------------------------------------
__global__ void pack_kernel(const float* __restrict__ s1, const float* __restrict__ rmax,
                            const float* __restrict__ rrsum,
                            const float* __restrict__ s2, const float* __restrict__ cmax,
                            const float* __restrict__ crsum,
                            float4* __restrict__ packRow, float4* __restrict__ packCol) {
  const int i = blockIdx.x * 256 + threadIdx.x;
  packRow[i] = make_float4(s1[i], rmax[i], rrsum[i], 0.f);
  packCol[i] = make_float4(s2[i], cmax[i], crsum[i], 0.f);
}

// ---------------------------------------------------------------------------
// Aggregation: out[m,f] = ELU( sum_k w(m,k) * Bh[k,f] ) via v_wmma_f32_16x16x32_f16
//   w(m,k) = exp( (adj>0 ? lrelu(s_m[m]+pk.x) : NEG_BIG) - pk.y ) * pk.z
// TRANSPOSED=0: m=i, k=j, adj[m,k] -> h_prime1 (column stats)
// TRANSPOSED=1: m=j, k=i, adj[k,m] -> h_prime2 (row stats)
// 128 threads (4 waves x one 16x16 N-subtile); K=64 per barrier round (2 WMMAs).
// All loops have uniform trip counts -> no divergent exec in the hot path.
// ---------------------------------------------------------------------------
template <bool TRANSPOSED>
__global__ void aggregate_kernel(const float* __restrict__ adj,
                                 const float* __restrict__ s_m,
                                 const float4* __restrict__ packK,
                                 const _Float16* __restrict__ Bh,
                                 float* __restrict__ out) {
  __shared__ u32 s_w32[16 * (WSTRIDE / 2)];   // weight tile halves [m][k]
  __shared__ u32 s_b32[64 * (WSTRIDE / 2)];   // B tile transposed halves [f][k]
  __shared__ float shm[16];
  _Float16* wt = reinterpret_cast<_Float16*>(s_w32);
  _Float16* bt = reinterpret_cast<_Float16*>(s_b32);

  const int tid  = threadIdx.x;
  const int lane = tid & 31;
  const int wv   = tid >> 5;
  const int m0   = blockIdx.x * 16;
  const int mrow = lane & 15;
  const int lh   = lane >> 4;
  const int n    = (wv << 4) + mrow;

  if (tid < 16) shm[tid] = s_m[m0 + tid];
  __syncthreads();

  v8f c = {};
  for (int k0 = 0; k0 < N_NODES; k0 += KCHUNK) {
    // (a) 16 x 64 f16 attention-weight tile (8 uniform steps, 128 thr)
    #pragma unroll
    for (int step = 0; step < 8; ++step) {
      int e = tid + step * 128;
      int m, k;
      if (!TRANSPOSED) { m = e >> 6; k = e & 63; }
      else             { m = e & 15; k = e >> 4; }
      size_t addr = TRANSPOSED ? ((size_t)(k0 + k) * N_NODES + (m0 + m))
                               : ((size_t)(m0 + m) * N_NODES + (k0 + k));
      float  aij = adj[addr];
      float4 pk  = packK[k0 + k];              // {s_k, kmax, 1/ksum}
      float sc   = shm[m] + pk.x;
      sc = sc > 0.f ? sc : ALPHA * sc;
      float att  = aij > 0.f ? sc : NEG_BIG;
      float w    = __expf(att - pk.y) * pk.z;
      wt[m * WSTRIDE + k] = (_Float16)w;
    }
    // (b) stage 64 x 64 f16 Wh tile transposed into LDS [f][k]
    {
      int kk = tid >> 1;                       // 0..63
      int f0 = (tid & 1) << 5;                 // 0 or 32
      union { uint4 q[4]; _Float16 h[32]; } u;
      const uint4* g = reinterpret_cast<const uint4*>(Bh + (size_t)(k0 + kk) * F_OUT + f0);
      u.q[0] = g[0]; u.q[1] = g[1]; u.q[2] = g[2]; u.q[3] = g[3];
      #pragma unroll
      for (int ff = 0; ff < 32; ++ff)
        bt[(f0 + ff) * WSTRIDE + kk] = u.h[ff];
    }
    __syncthreads();
    // (c) two 16x16x32 WMMAs over this K-chunk
    #pragma unroll
    for (int sub = 0; sub < 2; ++sub) {
      union { u32 u[8]; v16h v; } af, bf;
      const int abase = mrow * (WSTRIDE / 2) + sub * 16 + lh * 8;
      #pragma unroll
      for (int vv = 0; vv < 4; ++vv) {
        af.u[vv]     = s_w32[abase + vv];      // K = 2v,2v+1   (+16*lh, +32*sub)
        af.u[vv + 4] = s_w32[abase + 4 + vv];  // K = 8+2v,...
      }
      const int bbase = n * (WSTRIDE / 2) + sub * 16 + lh * 8;
      #pragma unroll
      for (int e = 0; e < 8; ++e)
        bf.u[e] = s_b32[bbase + e];            // K = 16*lh + 2e, 2e+1 (+32*sub)
      c = __builtin_amdgcn_wmma_f32_16x16x32_f16(false, af.v, false, bf.v,
                                                 (short)0, c, false, false);
    }
    __syncthreads();
  }
  // (d) fused ELU epilogue; C layout: VGPR r -> M = r + 8*lh, N = n
  #pragma unroll
  for (int r = 0; r < 8; ++r) {
    int m  = r + (lh ? 8 : 0);
    float x = c[r];
    x = x > 0.f ? x : (__expf(x) - 1.f);
    out[(size_t)(m0 + m) * F_OUT + n] = x;
  }
}

// ---------------------------------------------------------------------------
extern "C" void kernel_launch(void* const* d_in, const int* in_sizes, int n_in,
                              void* d_out, int out_size, void* d_ws, size_t ws_size,
                              hipStream_t stream) {
  const float* h1  = (const float*)d_in[0];
  const float* h2  = (const float*)d_in[1];
  const float* adj = (const float*)d_in[2];
  const float* W1  = (const float*)d_in[3];
  const float* W2  = (const float*)d_in[4];
  const float* a   = (const float*)d_in[5];
  float* out = (float*)d_out;

  // workspace layout (~8.7 MB)
  float* ws    = (float*)d_ws;
  float* Wh1   = ws;                               // 8192*64
  float* Wh2   = Wh1 + N_NODES * F_OUT;            // 8192*64
  float* s1    = Wh2 + N_NODES * F_OUT;            // 8192
  float* s2    = s1 + N_NODES;
  float* rmax  = s2 + N_NODES;
  float* rrsum = rmax + N_NODES;
  float* cmax  = rrsum + N_NODES;
  float* crsum = cmax + N_NODES;
  float* partM = crsum + N_NODES;                  // 32*8192
  float* partS = partM + 32 * N_NODES;             // 32*8192
  float4* packRow = (float4*)(partS + 32 * N_NODES);   // 8192 float4
  float4* packCol = packRow + N_NODES;                 // 8192 float4
  _Float16* Wh1h = (_Float16*)(packCol + N_NODES);     // 8192*64 halves
  _Float16* Wh2h = Wh1h + N_NODES * F_OUT;

  wh_kernel<<<dim3(N_NODES / 4, 2), 256, 0, stream>>>(h1, W1, h2, W2,
                                                      Wh1, Wh1h, Wh2, Wh2h);
  score_kernel<<<dim3(2 * N_NODES / 4), 256, 0, stream>>>(Wh1, Wh2, a, s1, s2);
  rowstats_kernel<<<dim3(N_NODES), 256, 0, stream>>>(adj, s1, s2, rmax, rrsum);
  colpart_kernel<<<dim3(N_NODES / 256, N_NODES / 256), 256, 0, stream>>>(adj, s1, s2,
                                                                         partM, partS);
  colreduce_kernel<<<dim3(N_NODES / 256), 256, 0, stream>>>(partM, partS, cmax, crsum);
  pack_kernel<<<dim3(N_NODES / 256), 256, 0, stream>>>(s1, rmax, rrsum,
                                                       s2, cmax, crsum,
                                                       packRow, packCol);

  // h_prime1 = attention2 @ Wh2  (K = columns j, column stats)
  aggregate_kernel<false><<<dim3(N_NODES / 16), 128, 0, stream>>>(
      adj, s1, packCol, Wh2h, out);
  // h_prime2 = attention1^T @ Wh1 (K = rows i, row stats), transposed adj access
  aggregate_kernel<true><<<dim3(N_NODES / 16), 128, 0, stream>>>(
      adj, s2, packRow, Wh1h, out + (size_t)N_NODES * F_OUT);
}
